// R4Wrapper_57294863729375
// MI455X (gfx1250) — matrix operand order repaired and checked by reading, not compiled, not run on hardware
//
#include <hip/hip_runtime.h>

// ---------------------------------------------------------------------------
// out[b,s,o] = sum_d FWHT(x)[b,s,d] * W[o,d] + bias[o]
//   M = 16384, N = 2048, K = 2048. Compute-bound (137 GFLOP vs ~285MB -> 12us
//   at 23.3 TB/s). Path: FWHT fp32 -> f16, GEMM via v_wmma_f32_16x16x32_f16,
//   staging via gfx1250 GLOBAL_LOAD_ASYNC_TO_LDS_B128 (ASYNCcnt), double-
//   buffered LDS, 64x64 wave tiles (16 WMMA : 16 ds_load per K-step).
// ---------------------------------------------------------------------------

typedef _Float16 v8h  __attribute__((ext_vector_type(8)));
typedef _Float16 v16h __attribute__((ext_vector_type(16)));
typedef float    v8f  __attribute__((ext_vector_type(8)));

#define DIM   2048
#define MROWS 16384
#define BM    256
#define BN    128
#define BK    32
#define LDSW  40               // padded f16 row stride (32 data + 8 pad)
#define STEPS (DIM / BK)       // 64

union Frag { v16h v; v8h h[2]; };

// async global->LDS, 16B per lane, ASYNCcnt-tracked
#define ASYNC_LD16(ldsoff, gaddr)                                     \
    asm volatile("global_load_async_to_lds_b128 %0, %1, off"          \
                 :: "v"(ldsoff), "v"(gaddr) : "memory")

// ---------------------------------------------------------------------------
// Kernel 1: per-row FWHT (strides 1..1024, matches reference butterfly),
// scale 1/sqrt(2048), convert to f16.
// ---------------------------------------------------------------------------
__global__ __launch_bounds__(256)
void hadamard_f16_kernel(const float* __restrict__ x, _Float16* __restrict__ xh) {
    __shared__ float s[DIM];
    const int row = blockIdx.x;
    const int t   = threadIdx.x;
    const float* __restrict__ xr = x + (size_t)row * DIM;

    const float4* x4 = (const float4*)xr;
    float4* s4 = (float4*)s;
    #pragma unroll
    for (int i = t; i < DIM / 4; i += 256) s4[i] = x4[i];
    __syncthreads();

    for (int st = 1; st < DIM; st <<= 1) {
        #pragma unroll
        for (int p = t; p < DIM / 2; p += 256) {
            int lo_bits = p & (st - 1);
            int i = ((p - lo_bits) << 1) | lo_bits;
            float a = s[i];
            float b = s[i + st];
            s[i]      = a + b;
            s[i + st] = a - b;
        }
        __syncthreads();
    }

    const float scale = 0.022097086912079610f;  // 1/sqrt(2048)
    _Float16* __restrict__ orow = xh + (size_t)row * DIM;
    v8h v;
    #pragma unroll
    for (int j = 0; j < 8; ++j) v[j] = (_Float16)(s[t * 8 + j] * scale);
    *(v8h*)(orow + t * 8) = v;
}

// ---------------------------------------------------------------------------
// Kernel 2: W fp32 -> f16 (row-major (o,d): already the B^T layout WMMA wants)
// ---------------------------------------------------------------------------
__global__ __launch_bounds__(256)
void wconv_kernel(const float* __restrict__ W, _Float16* __restrict__ Wh) {
    int i = (blockIdx.x * 256 + threadIdx.x) * 4;
    float4 w = *(const float4*)(W + i);
    union { _Float16 h[4]; uint2 u; } pk;
    pk.h[0] = (_Float16)w.x; pk.h[1] = (_Float16)w.y;
    pk.h[2] = (_Float16)w.z; pk.h[3] = (_Float16)w.w;
    *(uint2*)(Wh + i) = pk.u;
}

// ---------------------------------------------------------------------------
// Kernel 3: WMMA GEMM. 256 threads = 8 waves (4x2), block tile 256x128,
// wave tile 64x64 (4x4 accumulators), BK=32, double-buffered async LDS.
// ---------------------------------------------------------------------------
__global__ __launch_bounds__(256)
void gemm_kernel(const _Float16* __restrict__ A,   // MROWS x DIM
                 const _Float16* __restrict__ Wh,  // DIM x DIM (n,k)
                 const float*    __restrict__ bias,
                 float*          __restrict__ out) {
    __shared__ _Float16 sA[2][BM * LDSW];   // 2 x 20480 B
    __shared__ _Float16 sB[2][BN * LDSW];   // 2 x 10240 B

    const int m0   = blockIdx.y * BM;
    const int n0   = blockIdx.x * BN;
    const int tid  = threadIdx.x;
    const int lane = tid & 31;
    const int wave = tid >> 5;
    const int wm   = wave >> 1;       // 0..3 -> M offset wm*64
    const int wn   = wave & 1;        // 0..1 -> N offset wn*64
    const int lr   = lane & 15;
    const int lhi  = lane >> 4;

    // cooperative staging: thread -> (row = tid>>2, 16B chunk = tid&3)
    const int cchunk = tid & 3;
    const int crow   = tid >> 2;      // 0..63

    const uint32_t ROWSTEP = 64 * LDSW * 2;                 // 5120 B
    const uint32_t ASZ = BM * LDSW * 2;                     // 20480 B
    const uint32_t BSZ = BN * LDSW * 2;                     // 10240 B
    const uint64_t GROW = (uint64_t)64 * DIM * 2;           // 64 rows, bytes

    const uint32_t lA0 = (uint32_t)(size_t)&sA[0][crow * LDSW + cchunk * 8];
    const uint32_t lB0 = (uint32_t)(size_t)&sB[0][crow * LDSW + cchunk * 8];
    uint64_t gA0 = (uint64_t)(size_t)(A  + (size_t)(m0 + crow) * DIM + cchunk * 8);
    uint64_t gB0 = (uint64_t)(size_t)(Wh + (size_t)(n0 + crow) * DIM + cchunk * 8);

    auto issue = [&](int buf) {
        const uint32_t la = lA0 + (uint32_t)buf * ASZ;
        const uint32_t lb = lB0 + (uint32_t)buf * BSZ;
        ASYNC_LD16(la,               gA0);
        ASYNC_LD16(la + ROWSTEP,     gA0 + GROW);
        ASYNC_LD16(la + ROWSTEP * 2, gA0 + GROW * 2);
        ASYNC_LD16(la + ROWSTEP * 3, gA0 + GROW * 3);
        ASYNC_LD16(lb,               gB0);
        ASYNC_LD16(lb + ROWSTEP,     gB0 + GROW);
    };

    v8f acc[4][4] = {{}};

    issue(0);                                  // prologue: K-step 0 -> buf 0

    for (int ks = 0; ks < STEPS; ++ks) {
        asm volatile("s_wait_asynccnt 0x0" ::: "memory");
        __syncthreads();                       // all waves' tile data visible

        if (ks + 1 < STEPS) {                  // prefetch next K-step
            gA0 += BK * 2;
            gB0 += BK * 2;
            issue((ks + 1) & 1);
        }

        const _Float16* __restrict__ bufA = sA[ks & 1];
        const _Float16* __restrict__ bufB = sB[ks & 1];

        // A frag (16x32): lanes 0-15 -> K{0..7,16..23}; 16-31 -> K{8..15,24..31}
        Frag af[4];
        #pragma unroll
        for (int mt = 0; mt < 4; ++mt) {
            const int r = wm * 64 + mt * 16 + lr;
            af[mt].h[0] = *(const v8h*)(&bufA[r * LDSW + lhi * 8]);
            af[mt].h[1] = *(const v8h*)(&bufA[r * LDSW + 16 + lhi * 8]);
        }
        // B frag (32x16): lane%16 = N; lanes 0-15 -> K{0..7,16..23}; 16-31 -> K{8..15,24..31}
        Frag bf[4];
        #pragma unroll
        for (int nt = 0; nt < 4; ++nt) {
            const int r = wn * 64 + nt * 16 + lr;
            bf[nt].h[0] = *(const v8h*)(&bufB[r * LDSW + lhi * 8]);
            bf[nt].h[1] = *(const v8h*)(&bufB[r * LDSW + 16 + lhi * 8]);
        }

        #pragma unroll
        for (int mt = 0; mt < 4; ++mt)
            #pragma unroll
            for (int nt = 0; nt < 4; ++nt)
                acc[mt][nt] = __builtin_amdgcn_wmma_f32_16x16x32_f16(
                    false, af[mt].v, false, bf[nt].v,
                    (short)0, acc[mt][nt], false, false);
    }

    // epilogue: C/D layout (VGPR v -> M = v + 8*lhi, lane%16 -> N), add bias
    #pragma unroll
    for (int nt = 0; nt < 4; ++nt) {
        const int col = n0 + wn * 64 + nt * 16 + lr;
        const float bv = bias[col];
        #pragma unroll
        for (int mt = 0; mt < 4; ++mt) {
            const int mbase = m0 + wm * 64 + mt * 16 + lhi * 8;
            #pragma unroll
            for (int v = 0; v < 8; ++v)
                out[(size_t)(mbase + v) * DIM + col] = acc[mt][nt][v] + bv;
        }
    }
}

// ---------------------------------------------------------------------------
extern "C" void kernel_launch(void* const* d_in, const int* in_sizes, int n_in,
                              void* d_out, int out_size, void* d_ws, size_t ws_size,
                              hipStream_t stream) {
    const float* x    = (const float*)d_in[0];
    const float* W    = (const float*)d_in[1];
    const float* bias = (const float*)d_in[2];
    float* out        = (float*)d_out;

    _Float16* xh = (_Float16*)d_ws;                                     // 64 MB
    _Float16* Wh = (_Float16*)((char*)d_ws + (size_t)MROWS * DIM * 2);  // +8 MB

    hadamard_f16_kernel<<<MROWS, 256, 0, stream>>>(x, xh);
    wconv_kernel<<<(DIM * DIM) / (256 * 4), 256, 0, stream>>>(W, Wh);
    gemm_kernel<<<dim3(DIM / BN, MROWS / BM), 256, 0, stream>>>(xh, Wh, bias, out);
}
